// Mambablock_60533269069930
// MI455X (gfx1250) — compile-verified, matrix operand order checked by reading, not run on hardware
//
#include <hip/hip_runtime.h>
#include <hip/hip_bf16.h>
#include <math.h>

// ---------------------------------------------------------------------------
// Mamba block for MI455X (gfx1250, wave32).  All dense projections run through
// v_wmma_f32_16x16x32_f16; B tiles are streamed into double-buffered LDS with
// gfx1250 GLOBAL_LOAD_ASYNC_TO_LDS_B128; the selective scan is 3-phase chunked.
// ---------------------------------------------------------------------------

typedef __attribute__((ext_vector_type(16))) _Float16      v16h;
typedef __attribute__((ext_vector_type(8)))  float         v8f;
typedef __attribute__((ext_vector_type(4)))  unsigned int  u32x4;

union Frag { u32x4 q[2]; v16h h; };   // 32 bytes = one 16x32 f16 fragment/lane

#define L_SEQ   2048
#define D_MODEL 768
#define D_INNER 1536
#define D_STATE 16
#define DT_RANK 48
#define NCHUNK  64
#define CHLEN   32          // NCHUNK * CHLEN == L_SEQ
#define ROWP    40          // LDS row pitch in halves (bank-conflict-free b128)

// Exact pointer types for the async-to-LDS builtin:
// param0 = v4i addrspace(1)* (global), param1 = v4i addrspace(3)* (LDS).
typedef int v4i __attribute__((vector_size(16)));
typedef __attribute__((address_space(1))) v4i gv4i;
typedef __attribute__((address_space(3))) v4i lv4i;

// One 16-byte global -> LDS copy.  Prefer the CDNA5 async DMA path (ASYNCcnt,
// no VGPR round trip); fall back to load+store if the builtin is unavailable.
__device__ __forceinline__ void cp_b128(const _Float16* gsrc, _Float16* ldst) {
#if __has_builtin(__builtin_amdgcn_global_load_async_to_lds_b128)
  __builtin_amdgcn_global_load_async_to_lds_b128((gv4i*)gsrc, (lv4i*)ldst, 0, 0);
#else
  *reinterpret_cast<u32x4*>(ldst) = *reinterpret_cast<const u32x4*>(gsrc);
#endif
}

__device__ __forceinline__ void wait_async_all() {
  asm volatile("s_wait_asynccnt 0x0" ::: "memory");
}

// ---------------------------------------------------------------------------
// WMMA GEMM:  Y[M,N] = A[M,K] @ W[N,K]^T (+bias)(+res), f16 in / f32 out.
// Block = 256 threads = 8 waves; tile = 256 rows x 64 cols.
// Each wave owns 32 rows: 2 A fragments x 4 B fragments = 8 WMMAs per k-step.
// The 64x32 B tile is double-buffered in LDS and filled by async-to-LDS DMA
// (one b128 per thread) overlapped with the WMMA work on the previous tile.
// ---------------------------------------------------------------------------
__global__ __launch_bounds__(256)
void wmma_gemm(const _Float16* __restrict__ A, const _Float16* __restrict__ W,
               float* __restrict__ Y, const float* __restrict__ bias,
               const float* __restrict__ res, int M, int N, int K) {
  __shared__ _Float16 ldsB[2][64 * ROWP];

  const int lane = threadIdx.x & 31;
  const int wave = threadIdx.x >> 5;
  const int m0   = blockIdx.y * 256 + wave * 32;   // 2 row-fragments per wave
  const int nblk = blockIdx.x * 64;
  const int lrow = lane & 15;         // row within 16 (A: M, B: N)
  const int lhi  = lane >> 4;         // lane-group select
  const int kfr  = lhi * 8;           // K base of this lane group's halves

  // Cooperative B staging: thread -> (row nl, 16B chunk c) of the 64x32 tile.
  const int nl = threadIdx.x >> 2;    // 0..63
  const int c  = threadIdx.x & 3;     // 0..3
  const _Float16* wsrc = W + (size_t)(nblk + nl) * K + c * 8;
  _Float16* ldst[2] = { &ldsB[0][nl * ROWP + c * 8], &ldsB[1][nl * ROWP + c * 8] };

  v8f acc[2][4] = {};                 // [row frag][col frag] 16x16 f32 tiles

  const int nsteps = K / 32;
  cp_b128(wsrc, ldst[0]);             // prologue: tile 0 in flight

  for (int i = 0; i < nsteps; ++i) {
    wait_async_all();                 // my slice of tile i landed
    __syncthreads();                  // everyone's slice landed; buf free
    if (i + 1 < nsteps)               // stream tile i+1 while computing tile i
      cp_b128(wsrc + (size_t)(i + 1) * 32, ldst[(i + 1) & 1]);

    const _Float16* buf = ldsB[i & 1];

    // All 4 B fragments first (back-to-back ds_load_b128, conflict-free),
    // so the 8 WMMAs can issue without interleaved LDS waits.
    Frag bf[4];
#pragma unroll
    for (int f = 0; f < 4; ++f) {
      const _Float16* bp = buf + (f * 16 + lrow) * ROWP + kfr;
      bf[f].q[0] = *reinterpret_cast<const u32x4*>(bp);
      bf[f].q[1] = *reinterpret_cast<const u32x4*>(bp + 16);
    }

    // Two A fragments (rows m0, m0+16), gfx1250 16-bit A layout.
    Frag a0, a1;
    const _Float16* ap0 = A + (size_t)(m0 + lrow) * K + i * 32 + kfr;
    const _Float16* ap1 = ap0 + (size_t)16 * K;
    a0.q[0] = *reinterpret_cast<const u32x4*>(ap0);
    a0.q[1] = *reinterpret_cast<const u32x4*>(ap0 + 16);
    a1.q[0] = *reinterpret_cast<const u32x4*>(ap1);
    a1.q[1] = *reinterpret_cast<const u32x4*>(ap1 + 16);
    if (i + 1 < nsteps) {             // gfx1250 global_prefetch_b8
      __builtin_prefetch(ap0 + 32, 0, 1);
      __builtin_prefetch(ap1 + 32, 0, 1);
    }

#pragma unroll
    for (int f = 0; f < 4; ++f) {
      acc[0][f] = __builtin_amdgcn_wmma_f32_16x16x32_f16(
          false, a0.h, false, bf[f].h, (short)0, acc[0][f], false, false);
      acc[1][f] = __builtin_amdgcn_wmma_f32_16x16x32_f16(
          false, a1.h, false, bf[f].h, (short)0, acc[1][f], false, false);
    }
  }

  // Epilogue.  D layout: element v -> row +v + 8*lhi, col nblk + 16f + lrow.
#pragma unroll
  for (int r = 0; r < 2; ++r) {
#pragma unroll
    for (int f = 0; f < 4; ++f) {
      const int n = nblk + f * 16 + lrow;
#pragma unroll
      for (int v = 0; v < 8; ++v) {
        const int m = m0 + r * 16 + v + lhi * 8;
        float o = acc[r][f][v];
        if (bias) o += bias[n];
        const size_t off = (size_t)m * N + n;
        if (res) o += res[off];
        Y[off] = o;
      }
    }
  }
}

// --------------------------- LayerNorm -> f16 ------------------------------
__global__ __launch_bounds__(256)
void ln_f16(const float* __restrict__ x, const float* __restrict__ w,
            const float* __restrict__ b, _Float16* __restrict__ out, int C) {
  __shared__ float red[256];
  const int r = blockIdx.x, tid = threadIdx.x;
  const float* xr = x + (size_t)r * C;
  float s = 0.f;
  for (int c = tid; c < C; c += 256) s += xr[c];
  red[tid] = s; __syncthreads();
  for (int st = 128; st; st >>= 1) { if (tid < st) red[tid] += red[tid + st]; __syncthreads(); }
  const float mean = red[0] * (1.f / C);
  __syncthreads();
  float v = 0.f;
  for (int c = tid; c < C; c += 256) { float d = xr[c] - mean; v += d * d; }
  red[tid] = v; __syncthreads();
  for (int st = 128; st; st >>= 1) { if (tid < st) red[tid] += red[tid + st]; __syncthreads(); }
  const float inv = rsqrtf(red[0] * (1.f / C) + 1e-5f);
  for (int c = tid; c < C; c += 256)
    out[(size_t)r * C + c] = (_Float16)((xr[c] - mean) * inv * w[c] + b[c]);
}

// -------------------- weight f32 -> f16 with zero padding ------------------
__global__ __launch_bounds__(256)
void wcvt(const float* __restrict__ src, _Float16* __restrict__ dst,
          int N, int K, int Np, int Kp) {
  const int i = blockIdx.x * 256 + threadIdx.x;
  if (i >= Np * Kp) return;
  const int n = i / Kp, k = i % Kp;
  dst[i] = (n < N && k < K) ? (_Float16)src[(size_t)n * K + k] : (_Float16)0.f;
}

// --------------- depthwise causal conv(K=15) + SiLU, f32 + f16 -------------
__global__ __launch_bounds__(256)
void conv_silu(const float* __restrict__ xz, const float* __restrict__ cw,
               const float* __restrict__ cb, float* __restrict__ u32,
               _Float16* __restrict__ u16) {
  const int i = blockIdx.x * 256 + threadIdx.x;
  if (i >= L_SEQ * D_INNER) return;
  const int t = i / D_INNER, d = i % D_INNER;
  float acc = cb[d];
#pragma unroll
  for (int k = 0; k < 15; ++k) {
    const int ts = t - 14 + k;
    if (ts >= 0) acc += cw[d * 15 + k] * xz[(size_t)ts * (2 * D_INNER) + d];
  }
  const float s = acc / (1.f + __expf(-acc));
  u32[i] = s;
  u16[i] = (_Float16)s;
}

// -------- dt slice (cols 0..47 of dbc) -> f16, K padded 48 -> 64 -----------
__global__ __launch_bounds__(256)
void dtcast(const float* __restrict__ dbc, _Float16* __restrict__ dt16) {
  const int i = blockIdx.x * 256 + threadIdx.x;
  if (i >= L_SEQ * 64) return;
  const int t = i >> 6, k = i & 63;
  dt16[i] = (k < DT_RANK) ? (_Float16)dbc[(size_t)t * 128 + k] : (_Float16)0.f;
}

// ------------------------ delta = softplus(x + bias) -----------------------
__global__ __launch_bounds__(256)
void softplus_bias(float* __restrict__ io, const float* __restrict__ bias) {
  const int i = blockIdx.x * 256 + threadIdx.x;
  if (i >= L_SEQ * D_INNER) return;
  const float v = io[i] + bias[i % D_INNER];
  io[i] = (v > 20.f) ? v : log1pf(__expf(v));
}

// ---------------- scan phase 1: per-chunk transfer P and final F -----------
__global__ __launch_bounds__(256)
void scan_p1(const float* __restrict__ delta, const float* __restrict__ u32,
             const float* __restrict__ dbc, const float* __restrict__ A_log,
             float* __restrict__ PF) {
  const int g = blockIdx.x * 256 + threadIdx.x;      // g = c*D_INNER + d
  if (g >= NCHUNK * D_INNER) return;
  const int c = g / D_INNER, d = g % D_INNER;
  float A[D_STATE], h[D_STATE];
  float S = 0.f;
#pragma unroll
  for (int n = 0; n < D_STATE; ++n) { A[n] = -__expf(A_log[d * D_STATE + n]); h[n] = 0.f; }
  const int t0 = c * CHLEN;
  for (int tt = 0; tt < CHLEN; ++tt) {
    const int t = t0 + tt;
    const float dl = delta[(size_t)t * D_INNER + d];
    const float uu = u32[(size_t)t * D_INNER + d];
    S += dl;
    const float* Br = dbc + (size_t)t * 128 + DT_RANK;
#pragma unroll
    for (int n = 0; n < D_STATE; ++n) {
      const float a = __expf(dl * A[n]);
      h[n] = a * h[n] + dl * Br[n] * uu;
    }
  }
  float* o = PF + (size_t)g * 32;
#pragma unroll
  for (int n = 0; n < D_STATE; ++n) { o[n] = h[n]; o[16 + n] = __expf(A[n] * S); }
}

// ------------- scan phase 2: carry across chunks (tiny, 64 steps) ----------
__global__ __launch_bounds__(256)
void scan_p2(const float* __restrict__ PF, float* __restrict__ Hin) {
  const int g = blockIdx.x * 256 + threadIdx.x;      // g = d*16 + n
  if (g >= D_INNER * D_STATE) return;
  const int d = g >> 4, n = g & 15;
  float H = 0.f;
  for (int c = 0; c < NCHUNK; ++c) {
    const size_t base = (size_t)c * D_INNER + d;
    Hin[base * 16 + n] = H;
    H = PF[base * 32 + 16 + n] * H + PF[base * 32 + n];
  }
}

// ---------------- scan phase 3: replay chunk with true h_in, emit y --------
__global__ __launch_bounds__(256)
void scan_p3(const float* __restrict__ delta, const float* __restrict__ u32,
             const float* __restrict__ dbc, const float* __restrict__ A_log,
             const float* __restrict__ Hin, float* __restrict__ y) {
  const int g = blockIdx.x * 256 + threadIdx.x;
  if (g >= NCHUNK * D_INNER) return;
  const int c = g / D_INNER, d = g % D_INNER;
  float A[D_STATE], h[D_STATE];
#pragma unroll
  for (int n = 0; n < D_STATE; ++n) {
    A[n] = -__expf(A_log[d * D_STATE + n]);
    h[n] = Hin[(size_t)g * 16 + n];
  }
  const int t0 = c * CHLEN;
  for (int tt = 0; tt < CHLEN; ++tt) {
    const int t = t0 + tt;
    const float dl = delta[(size_t)t * D_INNER + d];
    const float uu = u32[(size_t)t * D_INNER + d];
    const float* Br = dbc + (size_t)t * 128 + DT_RANK;
    const float* Cr = Br + D_STATE;
    float yv = 0.f;
#pragma unroll
    for (int n = 0; n < D_STATE; ++n) {
      const float a = __expf(dl * A[n]);
      h[n] = a * h[n] + dl * Br[n] * uu;
      yv += h[n] * Cr[n];
    }
    y[(size_t)t * D_INNER + d] = yv;
  }
}

// ------------------- y_gated = (y + u*D) * silu(z)  ->  f16 ----------------
__global__ __launch_bounds__(256)
void gate_silu(const float* __restrict__ y, const float* __restrict__ u32,
               const float* __restrict__ Dp, const float* __restrict__ xz,
               _Float16* __restrict__ yg) {
  const int i = blockIdx.x * 256 + threadIdx.x;
  if (i >= L_SEQ * D_INNER) return;
  const int t = i / D_INNER, d = i % D_INNER;
  const float z = xz[(size_t)t * (2 * D_INNER) + D_INNER + d];
  const float g = z / (1.f + __expf(-z));
  yg[i] = (_Float16)((y[i] + u32[i] * Dp[d]) * g);
}

// -------------------------- exact GELU -> f16 ------------------------------
__global__ __launch_bounds__(256)
void gelu16(const float* __restrict__ in, _Float16* __restrict__ out, int n) {
  const int i = blockIdx.x * 256 + threadIdx.x;
  if (i >= n) return;
  const float v = in[i];
  out[i] = (_Float16)(0.5f * v * (1.f + erff(v * 0.70710678118f)));
}

// ===========================================================================
extern "C" void kernel_launch(void* const* d_in, const int* in_sizes, int n_in,
                              void* d_out, int out_size, void* d_ws, size_t ws_size,
                              hipStream_t stream) {
  (void)in_sizes; (void)n_in; (void)out_size; (void)ws_size;

  const float* x        = (const float*)d_in[0];
  const float* n1w      = (const float*)d_in[1];
  const float* n1b      = (const float*)d_in[2];
  const float* in_w     = (const float*)d_in[3];
  const float* in_b     = (const float*)d_in[4];
  const float* conv_w   = (const float*)d_in[5];
  const float* conv_b   = (const float*)d_in[6];
  const float* xp_w     = (const float*)d_in[7];
  const float* dt_w     = (const float*)d_in[8];
  const float* dt_b     = (const float*)d_in[9];
  const float* A_log    = (const float*)d_in[10];
  const float* Dp       = (const float*)d_in[11];
  const float* out_w    = (const float*)d_in[12];
  const float* out_b    = (const float*)d_in[13];
  const float* n2w      = (const float*)d_in[14];
  const float* n2b      = (const float*)d_in[15];
  const float* w1       = (const float*)d_in[16];
  const float* b1       = (const float*)d_in[17];
  const float* w2       = (const float*)d_in[18];
  const float* b2       = (const float*)d_in[19];
  float* out            = (float*)d_out;

  // ---- workspace bump allocator (256B aligned) ----
  char* ws = (char*)d_ws;
  size_t off = 0;
  auto alloc = [&](size_t bytes) -> void* {
    void* p = ws + off;
    off = (off + bytes + 255) & ~(size_t)255;
    return p;
  };
  _Float16* w_in16  = (_Float16*)alloc((size_t)2 * D_INNER * D_MODEL * 2);   // 3072x768
  _Float16* w_xp16  = (_Float16*)alloc((size_t)128 * D_INNER * 2);           // padded 80->128
  _Float16* w_dt16  = (_Float16*)alloc((size_t)D_INNER * 64 * 2);            // padded 48->64
  _Float16* w_out16 = (_Float16*)alloc((size_t)D_MODEL * D_INNER * 2);
  _Float16* w_m116  = (_Float16*)alloc((size_t)4 * D_MODEL * D_MODEL * 2);
  _Float16* w_m216  = (_Float16*)alloc((size_t)D_MODEL * 4 * D_MODEL * 2);
  _Float16* h16     = (_Float16*)alloc((size_t)L_SEQ * D_MODEL * 2);
  float*    xz      = (float*)   alloc((size_t)L_SEQ * 2 * D_INNER * 4);
  float*    u32b    = (float*)   alloc((size_t)L_SEQ * D_INNER * 4);
  _Float16* u16b    = (_Float16*)alloc((size_t)L_SEQ * D_INNER * 2);
  float*    dbc     = (float*)   alloc((size_t)L_SEQ * 128 * 4);
  _Float16* dt16    = (_Float16*)alloc((size_t)L_SEQ * 64 * 2);
  float*    delta   = (float*)   alloc((size_t)L_SEQ * D_INNER * 4);
  float*    PF      = (float*)   alloc((size_t)NCHUNK * D_INNER * 32 * 4);
  float*    Hin     = (float*)   alloc((size_t)NCHUNK * D_INNER * 16 * 4);
  float*    ybuf    = (float*)   alloc((size_t)L_SEQ * D_INNER * 4);
  _Float16* yg16    = (_Float16*)alloc((size_t)L_SEQ * D_INNER * 2);
  float*    xres    = (float*)   alloc((size_t)L_SEQ * D_MODEL * 4);
  _Float16* h2_16   = (_Float16*)alloc((size_t)L_SEQ * D_MODEL * 2);
  float*    m1      = (float*)   alloc((size_t)L_SEQ * 4 * D_MODEL * 4);
  _Float16* g16     = (_Float16*)alloc((size_t)L_SEQ * 4 * D_MODEL * 2);

  auto blocks = [](long n) { return (int)((n + 255) / 256); };
  auto cvt = [&](const float* s, _Float16* d, int N, int K, int Np, int Kp) {
    wcvt<<<blocks((long)Np * Kp), 256, 0, stream>>>(s, d, N, K, Np, Kp);
  };
  auto gemm = [&](const _Float16* A, const _Float16* W, float* Y,
                  const float* bias, const float* res, int M, int N, int K) {
    dim3 grid(N / 64, M / 256);
    wmma_gemm<<<grid, 256, 0, stream>>>(A, W, Y, bias, res, M, N, K);
  };

  // ---- f16 weight conversion (zero-padded to tile multiples) ----
  cvt(in_w,  w_in16,  2 * D_INNER, D_MODEL, 2 * D_INNER, D_MODEL);
  cvt(xp_w,  w_xp16,  DT_RANK + 2 * D_STATE, D_INNER, 128, D_INNER);
  cvt(dt_w,  w_dt16,  D_INNER, DT_RANK, D_INNER, 64);
  cvt(out_w, w_out16, D_MODEL, D_INNER, D_MODEL, D_INNER);
  cvt(w1,    w_m116,  4 * D_MODEL, D_MODEL, 4 * D_MODEL, D_MODEL);
  cvt(w2,    w_m216,  D_MODEL, 4 * D_MODEL, D_MODEL, 4 * D_MODEL);

  // ---- Mamba branch ----
  ln_f16<<<L_SEQ, 256, 0, stream>>>(x, n1w, n1b, h16, D_MODEL);
  gemm(h16, w_in16, xz, in_b, nullptr, L_SEQ, 2 * D_INNER, D_MODEL);
  conv_silu<<<blocks((long)L_SEQ * D_INNER), 256, 0, stream>>>(xz, conv_w, conv_b, u32b, u16b);
  gemm(u16b, w_xp16, dbc, nullptr, nullptr, L_SEQ, 128, D_INNER);
  dtcast<<<blocks((long)L_SEQ * 64), 256, 0, stream>>>(dbc, dt16);
  gemm(dt16, w_dt16, delta, nullptr, nullptr, L_SEQ, D_INNER, 64);
  softplus_bias<<<blocks((long)L_SEQ * D_INNER), 256, 0, stream>>>(delta, dt_b);

  // 3-phase chunked selective scan
  scan_p1<<<blocks((long)NCHUNK * D_INNER), 256, 0, stream>>>(delta, u32b, dbc, A_log, PF);
  scan_p2<<<blocks((long)D_INNER * D_STATE), 256, 0, stream>>>(PF, Hin);
  scan_p3<<<blocks((long)NCHUNK * D_INNER), 256, 0, stream>>>(delta, u32b, dbc, A_log, Hin, ybuf);

  gate_silu<<<blocks((long)L_SEQ * D_INNER), 256, 0, stream>>>(ybuf, u32b, Dp, xz, yg16);
  gemm(yg16, w_out16, xres, out_b, x, L_SEQ, D_MODEL, D_INNER);   // + residual x

  // ---- MLP branch ----
  ln_f16<<<L_SEQ, 256, 0, stream>>>(xres, n2w, n2b, h2_16, D_MODEL);
  gemm(h2_16, w_m116, m1, b1, nullptr, L_SEQ, 4 * D_MODEL, D_MODEL);
  gelu16<<<blocks((long)L_SEQ * 4 * D_MODEL), 256, 0, stream>>>(m1, g16, L_SEQ * 4 * D_MODEL);
  gemm(g16, w_m216, out, b2, xres, L_SEQ, D_MODEL, 4 * D_MODEL);  // + residual
}
